// SAB_LSTM_764504178981
// MI455X (gfx1250) — matrix operand order repaired and checked by reading, not compiled, not run on hardware
//
#include <hip/hip_runtime.h>
#include <math.h>

typedef __attribute__((ext_vector_type(16))) _Float16 v16h;
typedef __attribute__((ext_vector_type(8)))  float    v8f;

#define SEQ   512
#define BATCH 32
#define HID   512
#define G4    2048
#define TOPK  5
#define EPSV  1e-7f

#if __has_builtin(__builtin_amdgcn_tanhf)
#define TANHF(x) __builtin_amdgcn_tanhf(x)
#else
#define TANHF(x) tanhf(x)
#endif

static __device__ __forceinline__ float sigm_(float x) { return 1.0f / (1.0f + expf(-x)); }

// ---------------------------------------------------------------- f32 -> f16
__global__ void cvt_f32_f16(const float* __restrict__ src, _Float16* __restrict__ dst, int n) {
    int i = blockIdx.x * blockDim.x + threadIdx.x;
    int stride = gridDim.x * blockDim.x;
    for (; i < n; i += stride) dst[i] = (_Float16)src[i];
}

// ------------------------------------------------- generic WMMA GEMM + bias
// C[M][N] = A[M][K] * Bm[N][K]^T + bias0[n] (+ bias1[n])
// One wave computes a 16(M) x 64(N) strip. f16 in, f32 accumulate.
// k-loop manually double-buffered (unroll by 64) so fragment loads for k+32
// are in flight while the WMMAs for k execute.
union Frag { v16h v; float4 q[2]; };

__device__ __forceinline__ void ld_a(Frag& f, const _Float16* p) {
    f.q[0] = *(const float4*)(p);        // K = +0..7   (this lane's half)
    f.q[1] = *(const float4*)(p + 16);   // K = +16..23
}
__device__ __forceinline__ void ld_b(Frag& f, const _Float16* p) {
    f.q[0] = *(const float4*)(p);        // K = +0..7
    f.q[1] = *(const float4*)(p + 16);   // K = +16..23
}

__global__ __launch_bounds__(256) void wmma_gemm_bias(
    const _Float16* __restrict__ A,
    const _Float16* __restrict__ Bm,
    const float* __restrict__ bias0,
    const float* __restrict__ bias1,
    float* __restrict__ Cout,
    int M, int N, int K)   // K must be a multiple of 64
{
    const int wid  = (blockIdx.x * blockDim.x + threadIdx.x) >> 5;
    const int lane = threadIdx.x & 31;
    const int nStrips = N / 64;
    const int ms = wid / nStrips;
    const int ns = wid % nStrips;
    if (ms * 16 >= M) return;
    const int m  = lane & 15;
    const int hi = lane >> 4;

    v8f c[4] = {};
    const _Float16* ap = A + (size_t)(ms * 16 + m) * K + hi * 8;
    const _Float16* bp[4];
#pragma unroll
    for (int t = 0; t < 4; ++t)
        bp[t] = Bm + (size_t)(ns * 64 + t * 16 + m) * K + (hi << 3);

    Frag a0, a1, b0[4], b1[4];
    ld_a(a0, ap);
#pragma unroll
    for (int t = 0; t < 4; ++t) ld_b(b0[t], bp[t]);

    int kk = 0;
    for (; kk < K - 64; kk += 64) {
        // prefetch k+32 into buffer 1 while computing k with buffer 0
        ld_a(a1, ap + kk + 32);
#pragma unroll
        for (int t = 0; t < 4; ++t) ld_b(b1[t], bp[t] + kk + 32);
#pragma unroll
        for (int t = 0; t < 4; ++t)
            c[t] = __builtin_amdgcn_wmma_f32_16x16x32_f16(
                false, a0.v, false, b0[t].v, (short)0, c[t], false, false);
        // prefetch k+64 into buffer 0 while computing k+32 with buffer 1
        ld_a(a0, ap + kk + 64);
#pragma unroll
        for (int t = 0; t < 4; ++t) ld_b(b0[t], bp[t] + kk + 64);
#pragma unroll
        for (int t = 0; t < 4; ++t)
            c[t] = __builtin_amdgcn_wmma_f32_16x16x32_f16(
                false, a1.v, false, b1[t].v, (short)0, c[t], false, false);
    }
    // tail: kk == K-64
    ld_a(a1, ap + kk + 32);
#pragma unroll
    for (int t = 0; t < 4; ++t) ld_b(b1[t], bp[t] + kk + 32);
#pragma unroll
    for (int t = 0; t < 4; ++t)
        c[t] = __builtin_amdgcn_wmma_f32_16x16x32_f16(
            false, a0.v, false, b0[t].v, (short)0, c[t], false, false);
#pragma unroll
    for (int t = 0; t < 4; ++t)
        c[t] = __builtin_amdgcn_wmma_f32_16x16x32_f16(
            false, a1.v, false, b1[t].v, (short)0, c[t], false, false);

#pragma unroll
    for (int t = 0; t < 4; ++t) {
        int n = ns * 64 + t * 16 + m;
        float bv = (bias0 ? bias0[n] : 0.0f) + (bias1 ? bias1[n] : 0.0f);
#pragma unroll
        for (int r = 0; r < 8; ++r) {
            int row = ms * 16 + 8 * hi + r;
            Cout[(size_t)row * N + n] = c[t][r] + bv;
        }
    }
}

// ---------------------------------------------------------- SAB recurrence
// Single persistent workgroup (32 wave32s on one WGP). h carried in LDS (f16),
// c carried in registers. Per step: 32x2048x512 WMMA matmul + LSTM cell +
// incremental top-5 attention (h-term of the score cancels in s - delta, so
// aw has at most TOPK nonzeros and per-entry scores are computed only once).
__global__ __launch_bounds__(1024) void sab_recurrence(
    const float*    __restrict__ pre,     // [SEQ*32][2048] x@w_ih^T + b_ih + b_hh
    const _Float16* __restrict__ whh16,   // [2048][512]
    const float*    __restrict__ h0,
    const float*    __restrict__ c0,
    const float*    __restrict__ mem0,
    const float*    __restrict__ w_t,     // [1024]
    _Float16*       __restrict__ cat16,   // [SEQ*32][1024]  (h_seq | m_seq) f16
    float*          __restrict__ hf,      // [32*512]
    float*          __restrict__ cf,      // [32*512]
    float*          __restrict__ memf)    // [513*32][512]
{
    __shared__ _Float16 h16[32][512];      // 32 KB carried hidden state (f16)
    __shared__ float    part[32][32];      // per-wave score partials (deterministic)
    __shared__ float    top5s[32][TOPK];
    __shared__ int      top5i[32][TOPK];
    __shared__ float    minvS[32];
    __shared__ float    candw[32][TOPK];
    __shared__ int      candi[32][TOPK];

    const int tid  = threadIdx.x;
    const int w    = tid >> 5;
    const int lane = tid & 31;
    const int m    = lane & 15;
    const int hi   = lane >> 4;
    const int j    = w * 16 + m;               // hidden column owned by this lane
    const float wtm = w_t[512 + j];            // memory half of the scorer

    float creg[16];                             // cell state for 16 (b,j) pairs
    float p[16];

    // ---------------- prologue: load c0/h0, seed mem[0], score entry 0 ------
#pragma unroll
    for (int mt = 0; mt < 2; ++mt)
#pragma unroll
        for (int r = 0; r < 8; ++r) {
            int u = mt * 8 + r;
            int b = mt * 16 + 8 * hi + r;
            creg[u] = c0[b * 512 + j];
            h16[b][j] = (_Float16)h0[b * 512 + j];
            float mv = mem0[b * 512 + j];
            memf[(size_t)b * 512 + j] = mv;
            p[u] = TANHF(mv) * wtm;
        }
#pragma unroll
    for (int u = 0; u < 16; ++u) {
        p[u] += __shfl_xor(p[u], 1);
        p[u] += __shfl_xor(p[u], 2);
        p[u] += __shfl_xor(p[u], 4);
        p[u] += __shfl_xor(p[u], 8);
    }
    if (m == 0) {
#pragma unroll
        for (int u = 0; u < 16; ++u) {
            int b = (u >> 3) * 16 + 8 * hi + (u & 7);
            part[w][b] = p[u];
        }
    }
    __syncthreads();
    if (tid < 32) {
        int b = tid;
        float s0 = 0.0f;
        for (int w2 = 0; w2 < 32; ++w2) s0 += part[w2][b];
        top5s[b][0] = s0; top5i[b][0] = 0;
        for (int t = 1; t < TOPK; ++t) { top5s[b][t] = -3.0e38f; top5i[b][t] = 0; }
        minvS[b] = s0;
        // candidates for step 0 (cur = 1 <= TOPK -> delta = min + eps -> all aw 0)
        float delta = s0 + EPSV;
        float wsum = 0.0f, wr[TOPK];
        for (int t = 0; t < TOPK; ++t) { wr[t] = fmaxf(top5s[b][t] - delta, 0.0f); wsum += wr[t]; }
        float inv = 1.0f / (wsum + EPSV);
        for (int t = 0; t < TOPK; ++t) { candw[b][t] = wr[t] * inv; candi[b][t] = top5i[b][t]; }
    }
    __syncthreads();

    // ---------------- 512 sequential steps ---------------------------------
    for (int i = 0; i < SEQ; ++i) {
        // Warm the caches for this step's sequential-phase data while the
        // matmul runs: this step's gates_pre slab (256 KB) and the <=5
        // candidate memory rows (64 KB each). Lowers to global_prefetch_b8.
        {
            const float* pp = pre + (size_t)i * 32 * 2048;
            __builtin_prefetch(pp + tid * 64, 0, 0);
            __builtin_prefetch(pp + tid * 64 + 32, 0, 0);
#pragma unroll
            for (int t = 0; t < TOPK; ++t) {
                int id = candi[w][t];
                __builtin_prefetch(memf + ((size_t)id * 32 + w) * 512 + lane * 16, 0, 0);
            }
        }

        // phase 1: gates = h @ w_hh^T (wave w owns gate columns j = w*16..+15
        // for all four gates: n-tiles {q*32 + w}). Loads of a k-step are
        // grouped ahead of its 8 WMMAs (one wait per k-step).
        v8f Cg[4][2] = {};
#pragma unroll 2
        for (int kk = 0; kk < 512; kk += 32) {
            Frag a0, a1, bf[4];
            a0.q[0] = *(const float4*)&h16[m][kk + hi * 8];
            a0.q[1] = *(const float4*)&h16[m][kk + 16 + hi * 8];
            a1.q[0] = *(const float4*)&h16[16 + m][kk + hi * 8];
            a1.q[1] = *(const float4*)&h16[16 + m][kk + 16 + hi * 8];
#pragma unroll
            for (int q = 0; q < 4; ++q) {
                const _Float16* bptr = whh16 + (size_t)(q * 512 + w * 16 + m) * 512 + (hi << 3) + kk;
                bf[q].q[0] = *(const float4*)(bptr);
                bf[q].q[1] = *(const float4*)(bptr + 16);
            }
#pragma unroll
            for (int q = 0; q < 4; ++q) {
                Cg[q][0] = __builtin_amdgcn_wmma_f32_16x16x32_f16(
                    false, a0.v, false, bf[q].v, (short)0, Cg[q][0], false, false);
                Cg[q][1] = __builtin_amdgcn_wmma_f32_16x16x32_f16(
                    false, a1.v, false, bf[q].v, (short)0, Cg[q][1], false, false);
            }
        }
        __syncthreads();   // everyone done reading h16 before it is rewritten

        // phase 2: LSTM cell + apply attention (<=5 nonzero weights) per lane
#pragma unroll
        for (int mt = 0; mt < 2; ++mt) {
#pragma unroll
            for (int r = 0; r < 8; ++r) {
                int u = mt * 8 + r;
                int b = mt * 16 + 8 * hi + r;
                size_t base = (size_t)(i * 32 + b) * 2048 + j;
                float gi = Cg[0][mt][r] + pre[base];
                float gf = Cg[1][mt][r] + pre[base + 512];
                float gg = Cg[2][mt][r] + pre[base + 1024];
                float go = Cg[3][mt][r] + pre[base + 1536];
                float cn = sigm_(gf) * creg[u] + sigm_(gi) * TANHF(gg);
                creg[u] = cn;
                float hn = sigm_(go) * TANHF(cn);
                float mtv = 0.0f;
#pragma unroll
                for (int t = 0; t < TOPK; ++t) {
                    float wv = candw[b][t];
                    int   id = candi[b][t];
                    mtv += wv * memf[((size_t)id * 32 + b) * 512 + j];
                }
                float ho = hn + mtv;
                memf[((size_t)(i + 1) * 32 + b) * 512 + j] = ho;   // append entry cur=i+1
                size_t crow = (size_t)(i * 32 + b) * 1024;
                cat16[crow + j]       = (_Float16)ho;              // h_seq
                cat16[crow + 512 + j] = (_Float16)mtv;             // m_seq
                h16[b][j] = (_Float16)ho;                          // carry h
                p[u] = TANHF(ho) * wtm;                            // score of new entry
            }
        }
#pragma unroll
        for (int u = 0; u < 16; ++u) {
            p[u] += __shfl_xor(p[u], 1);
            p[u] += __shfl_xor(p[u], 2);
            p[u] += __shfl_xor(p[u], 4);
            p[u] += __shfl_xor(p[u], 8);
        }
        if (m == 0) {
#pragma unroll
            for (int u = 0; u < 16; ++u) {
                int b = (u >> 3) * 16 + 8 * hi + (u & 7);
                part[w][b] = p[u];
            }
        }
        __syncthreads();

        // phase 3: per-batch top-5 maintenance + candidate weights for step i+1
        if (tid < 32) {
            int b = tid;
            float sn = 0.0f;
            for (int w2 = 0; w2 < 32; ++w2) sn += part[w2][b];
            float s = sn; int id = i + 1;
#pragma unroll
            for (int t = 0; t < TOPK; ++t) {
                if (s > top5s[b][t]) {
                    float ts = top5s[b][t]; top5s[b][t] = s;  s  = ts;
                    int   ti = top5i[b][t]; top5i[b][t] = id; id = ti;
                }
            }
            float mn = fminf(minvS[b], sn); minvS[b] = mn;
            int count = i + 2;                               // valid entries for step i+1
            float delta = ((count <= TOPK) ? mn : top5s[b][TOPK - 1]) + EPSV;
            float wsum = 0.0f, wr[TOPK];
#pragma unroll
            for (int t = 0; t < TOPK; ++t) { wr[t] = fmaxf(top5s[b][t] - delta, 0.0f); wsum += wr[t]; }
            float inv = 1.0f / (wsum + EPSV);
#pragma unroll
            for (int t = 0; t < TOPK; ++t) { candw[b][t] = wr[t] * inv; candi[b][t] = top5i[b][t]; }
        }
        __syncthreads();
    }

    // epilogue: h_f (== last appended mem entry) and c_f
#pragma unroll
    for (int mt = 0; mt < 2; ++mt)
#pragma unroll
        for (int r = 0; r < 8; ++r) {
            int u = mt * 8 + r;
            int b = mt * 16 + 8 * hi + r;
            cf[b * 512 + j] = creg[u];
            hf[b * 512 + j] = memf[((size_t)SEQ * 32 + b) * 512 + j];
        }
}

// ---------------------------------------------------------------- launcher
extern "C" void kernel_launch(void* const* d_in, const int* in_sizes, int n_in,
                              void* d_out, int out_size, void* d_ws, size_t ws_size,
                              hipStream_t stream) {
    const float* x    = (const float*)d_in[0];
    const float* h0   = (const float*)d_in[1];
    const float* c0   = (const float*)d_in[2];
    const float* mem0 = (const float*)d_in[3];
    const float* w_ih = (const float*)d_in[4];
    const float* w_hh = (const float*)d_in[5];
    const float* b_ih = (const float*)d_in[6];
    const float* b_hh = (const float*)d_in[7];
    const float* w_t  = (const float*)d_in[8];
    const float* fc_w = (const float*)d_in[9];
    const float* fc_b = (const float*)d_in[10];

    // workspace layout (total 189,792,256 bytes)
    char* ws = (char*)d_ws;
    float*    gates_pre = (float*)   (ws);                 // 134217728 B
    _Float16* x16       = (_Float16*)(ws + 134217728);     //  16777216 B
    _Float16* cat16     = (_Float16*)(ws + 150994944);     //  33554432 B
    _Float16* wih16     = (_Float16*)(ws + 184549376);     //   2097152 B
    _Float16* whh16     = (_Float16*)(ws + 186646528);     //   2097152 B
    _Float16* fcw16     = (_Float16*)(ws + 188743680);     //   1048576 B

    // output layout: out | h_f | c_f | mem_f
    float* out  = (float*)d_out;
    float* hf   = out + 8388608;
    float* cf   = hf  + 16384;
    float* memf = cf  + 16384;

    cvt_f32_f16<<<1024, 256, 0, stream>>>(x,    x16,   SEQ * BATCH * HID);
    cvt_f32_f16<<<512,  256, 0, stream>>>(w_ih, wih16, G4 * HID);
    cvt_f32_f16<<<512,  256, 0, stream>>>(w_hh, whh16, G4 * HID);
    cvt_f32_f16<<<512,  256, 0, stream>>>(fc_w, fcw16, 512 * 1024);

    // K1: gates_pre = x @ w_ih^T + b_ih + b_hh   (M=16384, N=2048, K=512)
    wmma_gemm_bias<<<4096, 256, 0, stream>>>(x16, wih16, b_ih, b_hh,
                                             gates_pre, 16384, 2048, 512);
    // K2: sequential SAB-LSTM recurrence, one persistent workgroup
    sab_recurrence<<<1, 1024, 0, stream>>>(gates_pre, whh16, h0, c0, mem0, w_t,
                                           cat16, hf, cf, memf);
    // K3: out = [h_seq|m_seq] @ fc_w^T + fc_b    (M=16384, N=512, K=1024)
    wmma_gemm_bias<<<1024, 256, 0, stream>>>(cat16, fcw16, fc_b, nullptr,
                                             out, 16384, 512, 1024);
}
